// EGNN_80556406604617
// MI455X (gfx1250) — matrix operand order repaired and checked by reading, not compiled, not run on hardware
//
#include <hip/hip_runtime.h>
#include <hip/hip_bf16.h>

#define NN 10000
#define EE 160000
#define BB 16
#define IN_DIM 1280
#define HD 256
#define OUTD 512
#define LL 4
#define EDIM 4
#define K1 544            // 2H+ED=516 padded to multiple of 32
#define LNEPS 1e-5f

typedef __bf16 bf16;
typedef __attribute__((ext_vector_type(16))) __bf16 bf16x16;
typedef __attribute__((ext_vector_type(8)))  __bf16 bf16x8;
typedef __attribute__((ext_vector_type(8)))  float  f32x8;

static __device__ __forceinline__ f32x8 zero8() {
  f32x8 z;
#pragma unroll
  for (int i = 0; i < 8; ++i) z[i] = 0.0f;
  return z;
}

static __device__ __forceinline__ float silu_f(float v) {
  return v / (1.0f + __expf(-v));
}

// A fragment (16x32 bf16, MxK): lane row = lane&15; runs [hi*8, hi*8+8) and
// [16+hi*8, ...) of the 32-wide K chunk -> two contiguous 16B loads.
static __device__ __forceinline__ bf16x16 load_frag_a(const bf16* base, int stride, int kbase) {
  int lane = threadIdx.x & 31;
  int row  = lane & 15;
  int hi   = lane >> 4;
  const bf16* p = base + (size_t)row * stride + kbase + hi * 8;
  bf16x8 lo  = *(const bf16x8*)(p);
  bf16x8 hi8 = *(const bf16x8*)(p + 16);
  return __builtin_shufflevector(lo, hi8, 0,1,2,3,4,5,6,7,8,9,10,11,12,13,14,15);
}

// B fragment (32x16 bf16, KxN) from row-major weight W[o][k] (B[k][n]==W[n][k]):
// lane col = lane&15, K runs [hi*8, hi*8+8) and [16+hi*8, ...) -> two 16B loads.
static __device__ __forceinline__ bf16x16 load_frag_b(const bf16* base, int stride, int kbase, int ncol0) {
  int lane = threadIdx.x & 31;
  int col  = lane & 15;
  int hi   = lane >> 4;
  const bf16* p = base + (size_t)(ncol0 + col) * stride + kbase + hi * 8;
  bf16x8 lo  = *(const bf16x8*)(p);
  bf16x8 hi8 = *(const bf16x8*)(p + 16);
  return __builtin_shufflevector(lo, hi8, 0,1,2,3,4,5,6,7,8,9,10,11,12,13,14,15);
}

static __device__ __forceinline__ f32x8 wmma_bf16(bf16x16 a, bf16x16 b, f32x8 c) {
  return __builtin_amdgcn_wmma_f32_16x16x32_bf16(false, a, false, b, (short)0, c, false, false);
}

static __device__ __forceinline__ void atomicAddF(float* p, float v) {
  __hip_atomic_fetch_add(p, v, __ATOMIC_RELAXED, __HIP_MEMORY_SCOPE_AGENT);
}

// ---------------------------------------------------------------- utilities
__global__ void convert_pad_kernel(const float* __restrict__ src, bf16* __restrict__ dst,
                                   int rows, int scols, int dcols) {
  int i = blockIdx.x * blockDim.x + threadIdx.x;
  int total = rows * dcols;
  if (i >= total) return;
  int r = i / dcols, c = i - r * dcols;
  float v = (c < scols) ? src[(size_t)r * scols + c] : 0.0f;
  dst[i] = (bf16)v;
}

__global__ void zero_kernel(float* __restrict__ p, int n) {
  int i = blockIdx.x * blockDim.x + threadIdx.x;
  if (i < n) p[i] = 0.0f;
}

__global__ void deg_kernel(const int* __restrict__ ei, float* __restrict__ deg) {
  int e = blockIdx.x * blockDim.x + threadIdx.x;
  if (e < EE) atomicAddF(&deg[ei[EE + e]], 1.0f);
}

__global__ void cnt_kernel(const int* __restrict__ batch, float* __restrict__ cnt) {
  int n = blockIdx.x * blockDim.x + threadIdx.x;
  if (n < NN) atomicAddF(&cnt[batch[n]], 1.0f);
}

// --------------------------------------------------------- input projection
// x = silu(LN(nf @ inp_w.T + b));   16 nodes per block, 256 threads (8 waves)
__global__ void input_proj_kernel(const float* __restrict__ nf, const bf16* __restrict__ w,
                                  const float* __restrict__ b,
                                  const float* __restrict__ g, const float* __restrict__ be,
                                  float* __restrict__ xf, bf16* __restrict__ xb) {
  __shared__ bf16  sA[16 * IN_DIM];   // 40 KB
  __shared__ float sU[16 * HD];       // 16 KB
  __shared__ float sM[16], sV[16];
  int blk = blockIdx.x, t = threadIdx.x, wv = t >> 5;

  {  // stage + f32->bf16 convert: 16 threads per row, 80 halves each
    int r = t >> 4, c = t & 15;
    const float* srow = nf + (size_t)(blk * 16 + r) * IN_DIM + c * 80;
    bf16* drow = sA + r * IN_DIM + c * 80;
#pragma unroll 4
    for (int j = 0; j < 80; ++j) drow[j] = (bf16)srow[j];
  }
  __syncthreads();

  for (int tt = 0; tt < 2; ++tt) {
    int ct = wv * 2 + tt;
    f32x8 acc = zero8();
    for (int kc = 0; kc < IN_DIM / 32; ++kc)
      acc = wmma_bf16(load_frag_a(sA, IN_DIM, kc * 32),
                      load_frag_b(w, IN_DIM, kc * 32, ct * 16), acc);
    int lane = t & 31, col = ct * 16 + (lane & 15), hi = lane >> 4;
    float bias = b[col];
#pragma unroll
    for (int r = 0; r < 8; ++r) sU[(hi * 8 + r) * HD + col] = acc[r] + bias;
  }
  __syncthreads();

  if (t < 16) {
    float m = 0.f, v = 0.f;
    for (int j = 0; j < HD; ++j) m += sU[t * HD + j];
    m *= (1.0f / HD);
    for (int j = 0; j < HD; ++j) { float u = sU[t * HD + j] - m; v += u * u; }
    sM[t] = m; sV[t] = rsqrtf(v * (1.0f / HD) + LNEPS);
  }
  __syncthreads();
  {
    int r = t >> 4, c = t & 15;
    size_t n = (size_t)blk * 16 + r;
#pragma unroll 4
    for (int j = 0; j < 16; ++j) {
      int o = c * 16 + j;
      float u = (sU[r * HD + o] - sM[r]) * sV[r] * g[o] + be[o];
      u = silu_f(u);
      xf[n * HD + o] = u;
      xb[n * HD + o] = (bf16)u;
    }
  }
}

// ------------------------------------------------------------- edge kernel
// Fused 3-layer edge MLP for 32 edges/block; 128 threads (4 waves), each wave
// register-blocks 2 row-tiles x 4 col-tiles (8 WMMA accumulators) so each
// 32-wide K chunk issues 2 A-frag + 4 B-frag loads for 8 WMMAs.
__global__ void edge_kernel(const bf16* __restrict__ xb, const float* __restrict__ ea,
                            const int* __restrict__ ei,
                            const bf16* __restrict__ w1, const float* __restrict__ b1,
                            const bf16* __restrict__ w2, const float* __restrict__ b2,
                            const bf16* __restrict__ w3, const float* __restrict__ b3,
                            float* __restrict__ agg) {
  __shared__ bf16  sA[32 * K1];    // 34816 B  (reused as h2 after GEMM1)
  __shared__ bf16  sH[32 * HD];    // 16384 B  (h1)
  __shared__ float sInv[32];
  __shared__ int   sDst[32];
  bf16* sH2 = sA;                  // overlay: h2 (32x256) fits inside sA

  int blk = blockIdx.x, t = threadIdx.x, wv = t >> 5;   // wv in 0..3
  int cb = wv * 4;                                      // 4 col tiles per wave

  {  // stage m_in = [x[dst] | x[src] | edge_attr | 0-pad] as bf16, dword copies
    int r = t >> 2, c = t & 3;                  // 32 rows, 4 threads/row
    int e = blk * 32 + r;
    int src = ei[e], dst = ei[EE + e];
    if (c == 0) { sInv[r] = 1.0f / (ea[(size_t)e * 4] + 1e-8f); sDst[r] = dst; }
    unsigned int* dA = (unsigned int*)(sA + r * K1);
    const unsigned int* xd = (const unsigned int*)(xb + (size_t)dst * HD);
    const unsigned int* xs = (const unsigned int*)(xb + (size_t)src * HD);
#pragma unroll 4
    for (int i = 0; i < 68; ++i) {
      int uj = c * 68 + i;                      // 272 dwords per row
      unsigned int val;
      if (uj < 128) val = xd[uj];
      else if (uj < 256) val = xs[uj - 128];
      else if (uj < 258) {
        float f0 = ea[(size_t)e * 4 + (uj - 256) * 2];
        float f1 = ea[(size_t)e * 4 + (uj - 256) * 2 + 1];
        unsigned short u0 = __builtin_bit_cast(unsigned short, (bf16)f0);
        unsigned short u1 = __builtin_bit_cast(unsigned short, (bf16)f1);
        val = (unsigned int)u0 | ((unsigned int)u1 << 16);
      } else val = 0u;
      dA[uj] = val;
    }
  }
  __syncthreads();

  {  // GEMM1: (32 x 544) @ (544 x 256) -> silu -> sH
    f32x8 acc[2][4];
#pragma unroll
    for (int rr = 0; rr < 2; ++rr)
#pragma unroll
      for (int q = 0; q < 4; ++q) acc[rr][q] = zero8();
    for (int kc = 0; kc < K1 / 32; ++kc) {
      bf16x16 a0 = load_frag_a(sA,            K1, kc * 32);
      bf16x16 a1 = load_frag_a(sA + 16 * K1,  K1, kc * 32);
#pragma unroll
      for (int q = 0; q < 4; ++q) {
        bf16x16 bfr = load_frag_b(w1, K1, kc * 32, (cb + q) * 16);
        acc[0][q] = wmma_bf16(a0, bfr, acc[0][q]);
        acc[1][q] = wmma_bf16(a1, bfr, acc[1][q]);
      }
    }
    int lane = t & 31, col0 = lane & 15, hi = lane >> 4;
#pragma unroll
    for (int rr = 0; rr < 2; ++rr)
#pragma unroll
      for (int q = 0; q < 4; ++q) {
        int o = (cb + q) * 16 + col0;
        float bias = b1[o];
#pragma unroll
        for (int r = 0; r < 8; ++r) {
          int row = rr * 16 + hi * 8 + r;
          sH[row * HD + o] = (bf16)silu_f(acc[rr][q][r] + bias);
        }
      }
  }
  __syncthreads();

  {  // GEMM2: (32 x 256) @ (256 x 256) -> silu -> sH2 (overlay of sA)
    f32x8 acc[2][4];
#pragma unroll
    for (int rr = 0; rr < 2; ++rr)
#pragma unroll
      for (int q = 0; q < 4; ++q) acc[rr][q] = zero8();
    for (int kc = 0; kc < HD / 32; ++kc) {
      bf16x16 a0 = load_frag_a(sH,            HD, kc * 32);
      bf16x16 a1 = load_frag_a(sH + 16 * HD,  HD, kc * 32);
#pragma unroll
      for (int q = 0; q < 4; ++q) {
        bf16x16 bfr = load_frag_b(w2, HD, kc * 32, (cb + q) * 16);
        acc[0][q] = wmma_bf16(a0, bfr, acc[0][q]);
        acc[1][q] = wmma_bf16(a1, bfr, acc[1][q]);
      }
    }
    int lane = t & 31, col0 = lane & 15, hi = lane >> 4;
#pragma unroll
    for (int rr = 0; rr < 2; ++rr)
#pragma unroll
      for (int q = 0; q < 4; ++q) {
        int o = (cb + q) * 16 + col0;
        float bias = b2[o];
#pragma unroll
        for (int r = 0; r < 8; ++r) {
          int row = rr * 16 + hi * 8 + r;
          sH2[row * HD + o] = (bf16)silu_f(acc[rr][q][r] + bias);
        }
      }
  }
  __syncthreads();

  {  // GEMM3: (32 x 256) @ (256 x 256) -> *inv_dist -> atomic scatter into agg
    f32x8 acc[2][4];
#pragma unroll
    for (int rr = 0; rr < 2; ++rr)
#pragma unroll
      for (int q = 0; q < 4; ++q) acc[rr][q] = zero8();
    for (int kc = 0; kc < HD / 32; ++kc) {
      bf16x16 a0 = load_frag_a(sH2,           HD, kc * 32);
      bf16x16 a1 = load_frag_a(sH2 + 16 * HD, HD, kc * 32);
#pragma unroll
      for (int q = 0; q < 4; ++q) {
        bf16x16 bfr = load_frag_b(w3, HD, kc * 32, (cb + q) * 16);
        acc[0][q] = wmma_bf16(a0, bfr, acc[0][q]);
        acc[1][q] = wmma_bf16(a1, bfr, acc[1][q]);
      }
    }
    int lane = t & 31, col0 = lane & 15, hi = lane >> 4;
#pragma unroll
    for (int rr = 0; rr < 2; ++rr)
#pragma unroll
      for (int q = 0; q < 4; ++q) {
        int o = (cb + q) * 16 + col0;
        float bias = b3[o];
#pragma unroll
        for (int r = 0; r < 8; ++r) {
          int row = rr * 16 + hi * 8 + r;
          float v = (acc[rr][q][r] + bias) * sInv[row];
          atomicAddF(&agg[(size_t)sDst[row] * HD + o], v);
        }
      }
  }
}

// ------------------------------------------------------------- node kernel
// x = LN( silu([x | agg/deg] @ nw1.T + b1) @ nw2.T + b2 + x );  in-place on x
__global__ void node_kernel(const float* __restrict__ xf_in, const bf16* __restrict__ xb_in,
                            const float* __restrict__ agg, const float* __restrict__ deg,
                            const bf16* __restrict__ w1, const float* __restrict__ b1,
                            const bf16* __restrict__ w2, const float* __restrict__ b2,
                            const float* __restrict__ g, const float* __restrict__ be,
                            float* __restrict__ xf_out, bf16* __restrict__ xb_out) {
  __shared__ bf16  sA[16 * 2 * HD];   // 16 KB
  __shared__ bf16  sH[16 * HD];       // 8 KB
  __shared__ float sU[16 * HD];       // 16 KB
  __shared__ float sM[16], sV[16];
  int blk = blockIdx.x, t = threadIdx.x, wv = t >> 5;

  {  // stage concat [x_bf16 | bf16(agg/deg)]
    int r = t >> 4, c = t & 15;
    size_t n = (size_t)blk * 16 + r;
    float dg = deg[n]; if (dg < 1.0f) dg = 1.0f;
    float inv = 1.0f / dg;
#pragma unroll 4
    for (int j = 0; j < 32; ++j) {
      int k = c * 32 + j;
      bf16 v = (k < HD) ? xb_in[n * HD + k] : (bf16)(agg[n * HD + (k - HD)] * inv);
      sA[r * (2 * HD) + k] = v;
    }
  }
  __syncthreads();

  for (int tt = 0; tt < 2; ++tt) {  // GEMM1 K=512 -> silu -> sH
    int ct = wv * 2 + tt;
    f32x8 acc = zero8();
    for (int kc = 0; kc < (2 * HD) / 32; ++kc)
      acc = wmma_bf16(load_frag_a(sA, 2 * HD, kc * 32),
                      load_frag_b(w1, 2 * HD, kc * 32, ct * 16), acc);
    int lane = t & 31, col = ct * 16 + (lane & 15), hi = lane >> 4;
    float bias = b1[col];
#pragma unroll
    for (int r = 0; r < 8; ++r)
      sH[(hi * 8 + r) * HD + col] = (bf16)silu_f(acc[r] + bias);
  }
  __syncthreads();

  for (int tt = 0; tt < 2; ++tt) {  // GEMM2 K=256 -> + bias + residual -> sU
    int ct = wv * 2 + tt;
    f32x8 acc = zero8();
    for (int kc = 0; kc < HD / 32; ++kc)
      acc = wmma_bf16(load_frag_a(sH, HD, kc * 32),
                      load_frag_b(w2, HD, kc * 32, ct * 16), acc);
    int lane = t & 31, col = ct * 16 + (lane & 15), hi = lane >> 4;
    float bias = b2[col];
#pragma unroll
    for (int r = 0; r < 8; ++r) {
      int row = hi * 8 + r;
      size_t n = (size_t)blk * 16 + row;
      sU[row * HD + col] = acc[r] + bias + xf_in[n * HD + col];
    }
  }
  __syncthreads();

  if (t < 16) {
    float m = 0.f, v = 0.f;
    for (int j = 0; j < HD; ++j) m += sU[t * HD + j];
    m *= (1.0f / HD);
    for (int j = 0; j < HD; ++j) { float u = sU[t * HD + j] - m; v += u * u; }
    sM[t] = m; sV[t] = rsqrtf(v * (1.0f / HD) + LNEPS);
  }
  __syncthreads();
  {
    int r = t >> 4, c = t & 15;
    size_t n = (size_t)blk * 16 + r;
#pragma unroll 4
    for (int j = 0; j < 16; ++j) {
      int o = c * 16 + j;
      float u = (sU[r * HD + o] - sM[r]) * sV[r] * g[o] + be[o];
      xf_out[n * HD + o] = u;
      xb_out[n * HD + o] = (bf16)u;
    }
  }
}

// ------------------------------------------------------------ output stage
__global__ void out1_kernel(const bf16* __restrict__ xb, const bf16* __restrict__ w,
                            const float* __restrict__ b,
                            const float* __restrict__ g, const float* __restrict__ be,
                            bf16* __restrict__ hb) {
  __shared__ float sU[16 * HD];
  __shared__ float sM[16], sV[16];
  int blk = blockIdx.x, t = threadIdx.x, wv = t >> 5;
  const bf16* A = xb + (size_t)blk * 16 * HD;   // A fragments straight from global

  for (int tt = 0; tt < 2; ++tt) {
    int ct = wv * 2 + tt;
    f32x8 acc = zero8();
    for (int kc = 0; kc < HD / 32; ++kc)
      acc = wmma_bf16(load_frag_a(A, HD, kc * 32),
                      load_frag_b(w, HD, kc * 32, ct * 16), acc);
    int lane = t & 31, col = ct * 16 + (lane & 15), hi = lane >> 4;
    float bias = b[col];
#pragma unroll
    for (int r = 0; r < 8; ++r) sU[(hi * 8 + r) * HD + col] = acc[r] + bias;
  }
  __syncthreads();

  if (t < 16) {
    float m = 0.f, v = 0.f;
    for (int j = 0; j < HD; ++j) m += sU[t * HD + j];
    m *= (1.0f / HD);
    for (int j = 0; j < HD; ++j) { float u = sU[t * HD + j] - m; v += u * u; }
    sM[t] = m; sV[t] = rsqrtf(v * (1.0f / HD) + LNEPS);
  }
  __syncthreads();
  {
    int r = t >> 4, c = t & 15;
    size_t n = (size_t)blk * 16 + r;
#pragma unroll 4
    for (int j = 0; j < 16; ++j) {
      int o = c * 16 + j;
      float u = (sU[r * HD + o] - sM[r]) * sV[r] * g[o] + be[o];
      hb[n * HD + o] = (bf16)silu_f(u);
    }
  }
}

__global__ void out2_kernel(const bf16* __restrict__ hb, const bf16* __restrict__ w,
                            const float* __restrict__ b, const int* __restrict__ batch,
                            float* __restrict__ pool) {
  int blk = blockIdx.x, t = threadIdx.x, wv = t >> 5;
  const bf16* A = hb + (size_t)blk * 16 * HD;
#pragma unroll
  for (int q = 0; q < 4; ++q) {          // 32 col tiles (OUT=512), 4 per wave
    int ct = wv * 4 + q;
    f32x8 acc = zero8();
    for (int kc = 0; kc < HD / 32; ++kc)
      acc = wmma_bf16(load_frag_a(A, HD, kc * 32),
                      load_frag_b(w, HD, kc * 32, ct * 16), acc);
    int lane = t & 31, col = ct * 16 + (lane & 15), hi = lane >> 4;
    float bias = b[col];
#pragma unroll
    for (int r = 0; r < 8; ++r) {
      int row = hi * 8 + r;
      size_t n = (size_t)blk * 16 + row;
      atomicAddF(&pool[(size_t)batch[n] * OUTD + col], acc[r] + bias);
    }
  }
}

__global__ void pool_div_kernel(const float* __restrict__ pool, const float* __restrict__ cnt,
                                float* __restrict__ out) {
  int i = blockIdx.x * blockDim.x + threadIdx.x;
  if (i >= BB * OUTD) return;
  int b = i / OUTD;
  float c = cnt[b]; if (c < 1.0f) c = 1.0f;
  out[i] = pool[i] / c;
}

// ---------------------------------------------------------------- launcher
extern "C" void kernel_launch(void* const* d_in, const int* in_sizes, int n_in,
                              void* d_out, int out_size, void* d_ws, size_t ws_size,
                              hipStream_t stream) {
  (void)in_sizes; (void)n_in; (void)out_size; (void)ws_size;
  const float* nf    = (const float*)d_in[0];
  const float* ea    = (const float*)d_in[2];
  const int*   ei    = (const int*)d_in[3];
  const int*   batch = (const int*)d_in[4];
  const float* inp_w = (const float*)d_in[5];
  const float* inp_b = (const float*)d_in[6];
  const float* ilng  = (const float*)d_in[7];
  const float* ilnb  = (const float*)d_in[8];
  const float* ew1   = (const float*)d_in[9];
  const float* eb1   = (const float*)d_in[10];
  const float* ew2   = (const float*)d_in[11];
  const float* eb2   = (const float*)d_in[12];
  const float* ew3   = (const float*)d_in[13];
  const float* eb3   = (const float*)d_in[14];
  const float* nw1   = (const float*)d_in[15];
  const float* nb1   = (const float*)d_in[16];
  const float* nw2   = (const float*)d_in[17];
  const float* nb2   = (const float*)d_in[18];
  const float* lng   = (const float*)d_in[19];
  const float* lnb   = (const float*)d_in[20];
  const float* ow1   = (const float*)d_in[21];
  const float* ob1   = (const float*)d_in[22];
  const float* olng  = (const float*)d_in[23];
  const float* olnb  = (const float*)d_in[24];
  const float* ow2   = (const float*)d_in[25];
  const float* ob2   = (const float*)d_in[26];

  char* wbase = (char*)d_ws;
  size_t off = 0;
  auto carve = [&](size_t bytes) -> char* {
    char* p = wbase + off;
    off += (bytes + 255) & ~(size_t)255;
    return p;
  };
  float* xf   = (float*)carve((size_t)NN * HD * 4);
  bf16*  xb   = (bf16*) carve((size_t)NN * HD * 2);
  float* agg  = (float*)carve((size_t)NN * HD * 4);
  float* deg  = (float*)carve((size_t)NN * 4);
  bf16*  hb   = (bf16*) carve((size_t)NN * HD * 2);
  float* pool = (float*)carve((size_t)BB * OUTD * 4);
  float* cnt  = (float*)carve((size_t)BB * 4);
  bf16*  wInp = (bf16*)carve((size_t)HD * IN_DIM * 2);
  bf16*  wE1  = (bf16*)carve((size_t)LL * HD * K1 * 2);
  bf16*  wE2  = (bf16*)carve((size_t)LL * HD * HD * 2);
  bf16*  wE3  = (bf16*)carve((size_t)LL * HD * HD * 2);
  bf16*  wN1  = (bf16*)carve((size_t)LL * HD * 2 * HD * 2);
  bf16*  wN2  = (bf16*)carve((size_t)LL * HD * HD * 2);
  bf16*  wO1  = (bf16*)carve((size_t)HD * HD * 2);
  bf16*  wO2  = (bf16*)carve((size_t)OUTD * HD * 2);

  auto cvt = [&](const float* s, bf16* d, int rows, int sc, int dc) {
    int tot = rows * dc;
    convert_pad_kernel<<<(tot + 255) / 256, 256, 0, stream>>>(s, d, rows, sc, dc);
  };
  cvt(inp_w, wInp, HD, IN_DIM, IN_DIM);
  for (int i = 0; i < LL; ++i) {
    cvt(ew1 + (size_t)i * HD * (2 * HD + EDIM), wE1 + (size_t)i * HD * K1, HD, 2 * HD + EDIM, K1);
    cvt(ew2 + (size_t)i * HD * HD,     wE2 + (size_t)i * HD * HD,     HD, HD, HD);
    cvt(ew3 + (size_t)i * HD * HD,     wE3 + (size_t)i * HD * HD,     HD, HD, HD);
    cvt(nw1 + (size_t)i * HD * 2 * HD, wN1 + (size_t)i * HD * 2 * HD, HD, 2 * HD, 2 * HD);
    cvt(nw2 + (size_t)i * HD * HD,     wN2 + (size_t)i * HD * HD,     HD, HD, HD);
  }
  cvt(ow1, wO1, HD, HD, HD);
  cvt(ow2, wO2, OUTD, HD, HD);

  zero_kernel<<<(NN + 255) / 256, 256, 0, stream>>>(deg, NN);
  zero_kernel<<<(BB * OUTD + 255) / 256, 256, 0, stream>>>(pool, BB * OUTD);
  zero_kernel<<<1, 256, 0, stream>>>(cnt, BB);
  deg_kernel<<<(EE + 255) / 256, 256, 0, stream>>>(ei, deg);
  cnt_kernel<<<(NN + 255) / 256, 256, 0, stream>>>(batch, cnt);

  input_proj_kernel<<<NN / 16, 256, 0, stream>>>(nf, wInp, inp_b, ilng, ilnb, xf, xb);

  for (int i = 0; i < LL; ++i) {
    zero_kernel<<<(NN * HD + 255) / 256, 256, 0, stream>>>(agg, NN * HD);
    edge_kernel<<<EE / 32, 128, 0, stream>>>(xb, ea, ei,
        wE1 + (size_t)i * HD * K1, eb1 + i * HD,
        wE2 + (size_t)i * HD * HD, eb2 + i * HD,
        wE3 + (size_t)i * HD * HD, eb3 + i * HD, agg);
    node_kernel<<<NN / 16, 256, 0, stream>>>(xf, xb, agg, deg,
        wN1 + (size_t)i * HD * 2 * HD, nb1 + i * HD,
        wN2 + (size_t)i * HD * HD, nb2 + i * HD,
        lng + i * HD, lnb + i * HD, xf, xb);
  }

  out1_kernel<<<NN / 16, 256, 0, stream>>>(xb, wO1, ob1, olng, olnb, hb);
  out2_kernel<<<NN / 16, 256, 0, stream>>>(hb, wO2, ob2, batch, pool);
  pool_div_kernel<<<(BB * OUTD + 255) / 256, 256, 0, stream>>>(pool, cnt, (float*)d_out);
}
